// LSTMTagger_43937515438595
// MI455X (gfx1250) — matrix operand order repaired and checked by reading, not compiled, not run on hardware
//
#include <hip/hip_runtime.h>
#include <hip/hip_bf16.h>

// ---------------------------------------------------------------------------
// LSTM path-attention model for MI455X (gfx1250, wave32, WMMA bf16 path).
//   inputs (setup_inputs order):
//     0 embedding [100000,256] f32      5 w_lin [1,768] f32
//     1 w_ih [1024,256] f32             6 b_lin [1] f32
//     2 w_hh [1024,256] f32             7 paths [4096,8] int
//     3 b_ih [1024] f32                 8 user_id [1] int
//     4 b_hh [1024] f32                 9 item_id [1] int
//   output: [1] f32
// ---------------------------------------------------------------------------

typedef __attribute__((ext_vector_type(16))) __bf16 v16bf;
typedef __attribute__((ext_vector_type(8)))  float  v8f;

#define P_TOT 4096
#define LHOPS 8
#define DIM   256
#define HID   256
#define G4    1024
#define PT    16   // paths per workgroup

__device__ __forceinline__ unsigned short f2bf(float f) {
  unsigned int u = __float_as_uint(f);
  u += 0x7FFFu + ((u >> 16) & 1u);   // round-to-nearest-even
  return (unsigned short)(u >> 16);
}

__device__ __forceinline__ float sigm(float x) {
  return 1.0f / (1.0f + __expf(-x));
}

union FragBF {
  v16bf v;
  unsigned short s[16];
  uint4 q[2];
};

// A fragment: 16x32 bf16 tile from row-major LDS (row stride 256 elements).
// ISA 16-bit A layout: lanes 0-15 hold row M=lane, K = {0..7, 16..23};
// lanes 16-31 hold row M=lane-16, K shifted by +8.
__device__ __forceinline__ v16bf load_frag_a(const unsigned short* base, int kb, int lane) {
  const int row = lane & 15;
  const int off = kb + ((lane >> 4) << 3);
  const unsigned short* p = base + row * DIM + off;
  FragBF u;
  u.q[0] = *(const uint4*)(p);        // K = off+0..7
  u.q[1] = *(const uint4*)(p + 16);   // K = off+16..23
  return u.v;
}

// B fragment: 32x16 tile of W^T, i.e. 16 rows of row-major bf16 W [1024][256].
// lane holds column N = lane&15 (a row of W), K = {0..7, 16..23} + (lane>>4)*8.
__device__ __forceinline__ v16bf load_frag_b(const unsigned short* w, int col0, int kb, int lane) {
  const unsigned short* p = w + (size_t)(col0 + (lane & 15)) * DIM + kb + ((lane >> 4) << 3);
  FragBF u;
  u.q[0] = *(const uint4*)(p);
  u.q[1] = *(const uint4*)(p + 16);
  return u.v;
}

__global__ void cvt_bf16_kernel(const float* __restrict__ src,
                                unsigned short* __restrict__ dst, int n) {
  const int i = blockIdx.x * blockDim.x + threadIdx.x;
  if (i < n) dst[i] = f2bf(src[i]);
}

// One workgroup = 16 paths, full recurrence kept local. 8 waves each own a
// 128-column slab of the 1024 gate outputs.
__global__ void lstm_kernel(const float* __restrict__ emb,
                            const unsigned short* __restrict__ wih,   // [1024][256] bf16
                            const unsigned short* __restrict__ whh,   // [1024][256] bf16
                            const float* __restrict__ b_ih,
                            const float* __restrict__ b_hh,
                            const int* __restrict__ paths,
                            float* __restrict__ h_all,                // [4096][256]
                            float* __restrict__ logits) {             // [4096]
  extern __shared__ char smem[];
  float*          gates = (float*)smem;                          // [16][1024]  64 KB
  unsigned short* xbf   = (unsigned short*)(smem + 65536);       // [16][256]    8 KB
  unsigned short* hbf   = (unsigned short*)(smem + 65536 + 8192);// [16][256]    8 KB
  float*          bsum  = (float*)(smem + 65536 + 16384);        // [1024]       4 KB

  const int tid  = threadIdx.x;
  const int lane = tid & 31;
  const int wave = tid >> 5;
  const int p    = tid >> 4;    // 0..15 path within tile (gather / cell phases)
  const int s16  = tid & 15;
  const int p0   = blockIdx.x * PT;

  for (int i = tid; i < PT * HID; i += 256) hbf[i] = 0;
  for (int i = tid; i < G4; i += 256) bsum[i] = b_ih[i] + b_hh[i];

  float c[16];
#pragma unroll
  for (int j = 0; j < 16; ++j) c[j] = 0.0f;

  float lmax = -3.4e38f;

  for (int t = 0; t < LHOPS; ++t) {
    // ---- gather x_t for 16 paths -> bf16 LDS (64B contiguous per thread)
    {
      const int node = paths[(size_t)(p0 + p) * LHOPS + t];
      const float* src = emb + (size_t)node * DIM + s16 * 16;
      unsigned short* dst = xbf + p * DIM + s16 * 16;
#pragma unroll
      for (int i = 0; i < 16; ++i) dst[i] = f2bf(src[i]);
    }
    __syncthreads();

    // ---- gates[16][1024] = x_t @ W_ih^T + h @ W_hh^T  (WMMA bf16, f32 acc)
    for (int nt = 0; nt < 8; ++nt) {
      const int col0 = wave * 128 + nt * 16;
      v8f acc = {};
#pragma unroll
      for (int kb = 0; kb < DIM; kb += 32) {
        v16bf ah = load_frag_a(hbf, kb, lane);
        v16bf bh = load_frag_b(whh, col0, kb, lane);
        acc = __builtin_amdgcn_wmma_f32_16x16x32_bf16(false, ah, false, bh,
                                                      (short)0, acc, false, false);
        v16bf ax = load_frag_a(xbf, kb, lane);
        v16bf bx = load_frag_b(wih, col0, kb, lane);
        acc = __builtin_amdgcn_wmma_f32_16x16x32_bf16(false, ax, false, bx,
                                                      (short)0, acc, false, false);
      }
      // C/D layout: VGPR r, lanes 0-15 -> M=r, lanes 16-31 -> M=8+r
      const int n  = col0 + (lane & 15);
      const int mb = (lane >> 4) << 3;
#pragma unroll
      for (int r = 0; r < 8; ++r) gates[(mb + r) * G4 + n] = acc[r];
    }
    __syncthreads();

    // ---- LSTM cell (PyTorch gate order i,f,g,o); c lives in registers
#pragma unroll
    for (int j = 0; j < 16; ++j) {
      const int hid = s16 + (j << 4);
      const float gi = gates[p * G4 + hid]       + bsum[hid];
      const float gf = gates[p * G4 + 256 + hid] + bsum[256 + hid];
      const float gg = gates[p * G4 + 512 + hid] + bsum[512 + hid];
      const float go = gates[p * G4 + 768 + hid] + bsum[768 + hid];
      const float cs = sigm(gf) * c[j] + sigm(gi) * tanhf(gg);
      c[j] = cs;
      const float hh = sigm(go) * tanhf(cs);
      hbf[p * DIM + hid] = f2bf(hh);
      if (t == LHOPS - 1) {
        h_all[(size_t)(p0 + p) * HID + hid] = hh;
        lmax = fmaxf(lmax, hh);
      }
    }
    __syncthreads();
  }

  // ---- per-path max over hidden -> attention logits
  gates[tid] = lmax;
  __syncthreads();
  if (tid < PT) {
    float m = -3.4e38f;
#pragma unroll
    for (int q = 0; q < 16; ++q) m = fmaxf(m, gates[tid * 16 + q]);
    logits[p0 + tid] = m;
  }
}

// One workgroup: softmax over 4096 path logits, weighted hidden sum, final MLP.
__global__ void finalize_kernel(const float* __restrict__ emb,
                                const float* __restrict__ w_lin,
                                const float* __restrict__ b_lin,
                                const int* __restrict__ user_id,
                                const int* __restrict__ item_id,
                                const float* __restrict__ h_all,
                                const float* __restrict__ logits,
                                float* __restrict__ out) {
  __shared__ float red[256];
  const int tid = threadIdx.x;

  float m = -3.4e38f;
  for (int i = 0; i < 16; ++i) m = fmaxf(m, logits[tid * 16 + i]);
  red[tid] = m;
  __syncthreads();
  for (int s = 128; s > 0; s >>= 1) {
    if (tid < s) red[tid] = fmaxf(red[tid], red[tid + s]);
    __syncthreads();
  }
  const float gmax = red[0];
  __syncthreads();

  float z = 0.0f;
  for (int i = 0; i < 16; ++i) z += __expf(logits[tid * 16 + i] - gmax);
  red[tid] = z;
  __syncthreads();
  for (int s = 128; s > 0; s >>= 1) {
    if (tid < s) red[tid] += red[tid + s];
    __syncthreads();
  }
  const float Z = red[0];
  __syncthreads();

  // path_emb[tid] = sum_p h[p][tid] * softmax_p   (coalesced across threads)
  float acc = 0.0f;
  for (int pp = 0; pp < P_TOT; ++pp)
    acc += h_all[(size_t)pp * HID + tid] * __expf(logits[pp] - gmax);
  const float pe = acc / Z;

  const int uid = user_id[0];
  const int iid = item_id[0];
  const float contrib = emb[(size_t)uid * DIM + tid] * w_lin[tid]
                      + emb[(size_t)iid * DIM + tid] * w_lin[DIM + tid]
                      + pe * w_lin[2 * DIM + tid];
  red[tid] = contrib;
  __syncthreads();
  for (int s = 128; s > 0; s >>= 1) {
    if (tid < s) red[tid] += red[tid + s];
    __syncthreads();
  }
  if (tid == 0) out[0] = 1.0f / (1.0f + __expf(-(red[0] + b_lin[0])));
}

extern "C" void kernel_launch(void* const* d_in, const int* in_sizes, int n_in,
                              void* d_out, int out_size, void* d_ws, size_t ws_size,
                              hipStream_t stream) {
  const float* emb   = (const float*)d_in[0];
  const float* w_ih  = (const float*)d_in[1];
  const float* w_hh  = (const float*)d_in[2];
  const float* b_ih  = (const float*)d_in[3];
  const float* b_hh  = (const float*)d_in[4];
  const float* w_lin = (const float*)d_in[5];
  const float* b_lin = (const float*)d_in[6];
  const int*   paths = (const int*)d_in[7];
  const int*   uid   = (const int*)d_in[8];
  const int*   iid   = (const int*)d_in[9];

  char* ws = (char*)d_ws;
  float* h_all  = (float*)ws;                                         // 4 MB
  float* logits = (float*)(ws + (size_t)P_TOT * HID * sizeof(float)); // 16 KB
  unsigned short* wihbf =
      (unsigned short*)(ws + (size_t)P_TOT * HID * sizeof(float) + P_TOT * sizeof(float));
  unsigned short* whhbf = wihbf + (size_t)G4 * DIM;                   // 2x 512 KB

  const int wn = G4 * DIM;  // 262144 weights each
  cvt_bf16_kernel<<<(wn + 255) / 256, 256, 0, stream>>>(w_ih, wihbf, wn);
  cvt_bf16_kernel<<<(wn + 255) / 256, 256, 0, stream>>>(w_hh, whhbf, wn);

  const int smem_bytes = 65536 + 8192 + 8192 + 4096;  // 84 KB (CDNA5: 320 KB/WGP)
  hipFuncSetAttribute((const void*)lstm_kernel,
                      hipFuncAttributeMaxDynamicSharedMemorySize, smem_bytes);
  lstm_kernel<<<P_TOT / PT, 256, smem_bytes, stream>>>(
      emb, wihbf, whhbf, b_ih, b_hh, paths, h_all, logits);

  finalize_kernel<<<1, 256, 0, stream>>>(emb, w_lin, b_lin, uid, iid,
                                         h_all, logits, (float*)d_out);
}
